// AnatomicalMaskedLinear_9758165696678
// MI455X (gfx1250) — compile-verified, matrix operand order checked by reading, not run on hardware
//
#include <hip/hip_runtime.h>
#include <hip/hip_bf16.h>

// ---------------------------------------------------------------------------
// AnatomicalMaskedLinear for MI455X (gfx1250, wave32, WMMA bf16)
//   y[b, n_out, :] = sum_{n_in : adj[n_out][n_in]} x[b, n_in, :] @ W_block^T + bias
// Block-sparse GEMM, bf16 WMMA (16x16x32) with f32 accumulation.
// ---------------------------------------------------------------------------

#define NUM_NODES 21
#define IN_F      128
#define OUT_F     256
#define BATCH     4096
#define KDIM      (NUM_NODES * IN_F)   // 2688
#define NDIM      (NUM_NODES * OUT_F)  // 5376

#define BM 128   // batch-tile rows
#define BN 128   // output-feature tile cols
#define BK 128   // one input node's feature width

typedef __attribute__((ext_vector_type(16))) __bf16 v16bf;
typedef __attribute__((ext_vector_type(8)))  __bf16 v8bf;
typedef __attribute__((ext_vector_type(4)))  __bf16 v4bf;
typedef __attribute__((ext_vector_type(8)))  float  v8f;
typedef __attribute__((ext_vector_type(4)))  float  v4f;

__global__ __launch_bounds__(256)
void AnatomicalMaskedLinear_wmma_bf16(const float* __restrict__ x,
                                      const float* __restrict__ w,
                                      const float* __restrict__ bias,
                                      const int*   __restrict__ adj,
                                      float*       __restrict__ y)
{
    __shared__ __bf16 lA[BM * BK];   // x tile     (rows = batch,  cols = K)
    __shared__ __bf16 lB[BN * BK];   // W tile     (rows = outfeat, cols = K)

    const int tid   = threadIdx.x;
    const int lane  = tid & 31;
    const int wave  = tid >> 5;
    const int waveM = wave & 3;          // 0..3 -> 32-row band
    const int waveN = wave >> 2;         // 0..1 -> 64-col band

    const int rowBase = blockIdx.x * BM;                         // batch row
    const int nOut    = blockIdx.y >> 1;                         // 0..20
    const int colBase = nOut * OUT_F + (blockIdx.y & 1) * BN;    // global out col

    // ISA A/B fragment addressing: lane L holds matrix row (L & 15);
    // lanes 0-15 take K {0..7, 16..23}, lanes 16-31 take K {8..15, 24..31}.
    const int lr      = lane & 15;
    const int halfSel = (lane >> 4) & 1;

    v8f acc[2][4];
#pragma unroll
    for (int i = 0; i < 2; ++i)
#pragma unroll
        for (int j = 0; j < 4; ++j)
            acc[i][j] = (v8f){0.f, 0.f, 0.f, 0.f, 0.f, 0.f, 0.f, 0.f};

    for (int nin = 0; nin < NUM_NODES; ++nin) {
        if (adj[nOut * NUM_NODES + nin] == 0) continue;   // masked block: skip

        const float* gA = x + (size_t)rowBase * KDIM + (size_t)nin * IN_F;
        const float* gB = w + (size_t)colBase * KDIM + (size_t)nin * IN_F;

        // Hint the next node's columns toward L2 while we work on this one.
        __builtin_prefetch(gA + IN_F, 0, 1);
        __builtin_prefetch(gB + IN_F, 0, 1);

        // ---- stage both 128x128 f32 tiles into LDS as bf16 ----
        // 4096 float4 chunks per tile / 256 threads = 16 chunks each.
#pragma unroll
        for (int i = 0; i < 16; ++i) {
            const int c  = tid + i * 256;
            const int r  = c >> 5;          // 0..127
            const int c4 = (c & 31) << 2;   // 0,4,...,124

            v4f a = *(const v4f*)(gA + (size_t)r * KDIM + c4);
            v4f b = *(const v4f*)(gB + (size_t)r * KDIM + c4);

            v4bf pa, pb;
            pa.x = (__bf16)a.x; pa.y = (__bf16)a.y;
            pa.z = (__bf16)a.z; pa.w = (__bf16)a.w;
            pb.x = (__bf16)b.x; pb.y = (__bf16)b.y;
            pb.z = (__bf16)b.z; pb.w = (__bf16)b.w;

            *(v4bf*)&lA[r * BK + c4] = pa;
            *(v4bf*)&lB[r * BK + c4] = pb;
        }
        __syncthreads();

        // ---- 4 K-steps of 32, 8 WMMAs each ----
#pragma unroll
        for (int kk = 0; kk < BK; kk += 32) {
            v16bf afrag[2];
#pragma unroll
            for (int i = 0; i < 2; ++i) {
                const int row = waveM * 32 + i * 16 + lr;
                const int kb  = kk + halfSel * 8;
                v8bf lo = *(const v8bf*)&lA[row * BK + kb];
                v8bf hi = *(const v8bf*)&lA[row * BK + kb + 16];
#pragma unroll
                for (int e = 0; e < 8; ++e) {
                    afrag[i][e]     = lo[e];
                    afrag[i][e + 8] = hi[e];
                }
            }

            v16bf bfrag[4];
#pragma unroll
            for (int j = 0; j < 4; ++j) {
                const int row = waveN * 64 + j * 16 + lr;   // W row == out col
                const int kb  = kk + halfSel * 8;
                v8bf lo = *(const v8bf*)&lB[row * BK + kb];
                v8bf hi = *(const v8bf*)&lB[row * BK + kb + 16];
#pragma unroll
                for (int e = 0; e < 8; ++e) {
                    bfrag[j][e]     = lo[e];
                    bfrag[j][e + 8] = hi[e];
                }
            }

#pragma unroll
            for (int i = 0; i < 2; ++i)
#pragma unroll
                for (int j = 0; j < 4; ++j)
                    acc[i][j] = __builtin_amdgcn_wmma_f32_16x16x32_bf16(
                        /*neg_a=*/false, afrag[i],
                        /*neg_b=*/false, bfrag[j],
                        /*c_mod=*/(short)0, acc[i][j],
                        /*reuse_a=*/false, /*reuse_b=*/false);
        }
        __syncthreads();
    }

    // ---- epilogue: C/D layout -> global, + bias ----
    // VGPR v of a 16x16 f32 accumulator: M = v + 8*(lane>=16), N = lane%16.
#pragma unroll
    for (int i = 0; i < 2; ++i) {
#pragma unroll
        for (int j = 0; j < 4; ++j) {
            const int n  = colBase + waveN * 64 + j * 16 + lr;
            const float bv = bias[n];
#pragma unroll
            for (int v = 0; v < 8; ++v) {
                const int m = rowBase + waveM * 32 + i * 16 + v + halfSel * 8;
                y[(size_t)m * NDIM + n] = acc[i][j][v] + bv;
            }
        }
    }
}

extern "C" void kernel_launch(void* const* d_in, const int* in_sizes, int n_in,
                              void* d_out, int out_size, void* d_ws, size_t ws_size,
                              hipStream_t stream) {
    (void)in_sizes; (void)n_in; (void)out_size; (void)d_ws; (void)ws_size;
    const float* x    = (const float*)d_in[0];   // (4096, 21, 128) f32
    const float* w    = (const float*)d_in[1];   // (5376, 2688)    f32
    const float* bias = (const float*)d_in[2];   // (5376,)         f32
    const int*   adj  = (const int*)d_in[3];     // (21, 21)        i32
    float*       y    = (float*)d_out;           // (4096, 21, 256) f32

    dim3 grid(BATCH / BM, NUM_NODES * (OUT_F / BN));   // (32, 42)
    dim3 block(256);
    AnatomicalMaskedLinear_wmma_bf16<<<grid, block, 0, stream>>>(x, w, bias, adj, y);
}
